// Seq2Seq_3195455668279
// MI455X (gfx1250) — matrix operand order repaired and checked by reading, not compiled, not run on hardware
//
#include <hip/hip_runtime.h>
#include <hip/hip_bf16.h>
#include <math.h>

typedef __attribute__((ext_vector_type(16))) _Float16 v16h;
typedef __attribute__((ext_vector_type(8)))  _Float16 v8h;
typedef __attribute__((ext_vector_type(8)))  float    v8f;
typedef _Float16 f16;

#define DEVINL __device__ __forceinline__

#define BB 32
#define SS 256
#define EE 512
#define HD 256
#define HH 512
#define VV 32000
#define TT 256

// ---- WMMA fragment helpers (CDNA5 16x16x32 f16, wave32 layouts) ----

// A fragment: 16x32 f16 tile of row-major X (leading dim ld), tile origin (m0,k0).
// Lane L: row m0+(L&15); 16 contiguous K values X[m][k0+16*(L>>4) .. +16).
DEVINL v16h load_a16(const f16* X, long ld, int m0, int k0, int lane) {
  int m = m0 + (lane & 15);
  const f16* p = X + (long)m * ld + k0 + ((lane >> 4) << 4);
  v8h lo = *(const v8h*)p;
  v8h hi = *(const v8h*)(p + 8);
  v16h a;
#pragma unroll
  for (int i = 0; i < 8; ++i) { a[i] = lo[i]; a[i + 8] = hi[i]; }
  return a;
}

// B fragment from pre-converted f16 weight W [N,K] row-major (computes x@W.T).
// Lane L: col n0+(L&15), 16 contiguous K values from k0+16*(L>>4): two 16B loads.
DEVINL v16h load_b_h16(const f16* W, long ldw, int n0, int k0, int lane) {
  const f16* p = W + (long)(n0 + (lane & 15)) * ldw + k0 + ((lane >> 4) << 4);
  v8h lo = *(const v8h*)p;
  v8h hi = *(const v8h*)(p + 8);
  v16h b;
#pragma unroll
  for (int i = 0; i < 8; ++i) { b[i] = lo[i]; b[i + 8] = hi[i]; }
  return b;
}

// B fragment converted from f32 weights on the fly (used only by the one-shot generator).
DEVINL v16h load_b_w32(const float* W, long ldw, int n0, int k0, int lane) {
  const float* p = W + (long)(n0 + (lane & 15)) * ldw + k0 + ((lane >> 4) << 4);
  const float4* q = (const float4*)p;
  v16h b;
#pragma unroll
  for (int i = 0; i < 4; ++i) {
    float4 v = q[i];
    b[4 * i + 0] = (f16)v.x; b[4 * i + 1] = (f16)v.y;
    b[4 * i + 2] = (f16)v.z; b[4 * i + 3] = (f16)v.w;
  }
  return b;
}

DEVINL v8f wmma_f16(v16h a, v16h b, v8f c) {
  return __builtin_amdgcn_wmma_f32_16x16x32_f16(false, a, false, b, (short)0, c,
                                                false, false);
}

DEVINL v8f vzero8() {
  v8f z;
#pragma unroll
  for (int i = 0; i < 8; ++i) z[i] = 0.f;
  return z;
}

DEVINL float sigmoidf_(float x) { return 1.0f / (1.0f + __expf(-x)); }

// ---- generic f32 -> f16 weight pre-conversion ----
__global__ void cvt_f32_f16_kernel(const float* __restrict__ in,
                                   f16* __restrict__ out, int n) {
  for (int i = blockIdx.x * 256 + threadIdx.x; i < n; i += gridDim.x * 256)
    out[i] = (f16)in[i];
}

// ---- src embedding gather (f32 table -> f16 activations) ----
__global__ void embed_src_kernel(const int* __restrict__ src,
                                 const float* __restrict__ emb,
                                 f16* __restrict__ xout, int n) {
  int i = blockIdx.x * 256 + threadIdx.x;
  if (i < n) {
    int tok = src[i >> 9];  // E = 512
    xout[i] = (f16)emb[(long)tok * EE + (i & 511)];
  }
}

// ---- one bidirectional encoder layer: block 0 = fwd, block 1 = bwd ----
// 512 threads = 16 waves (4/SIMD for latency hiding); wave w owns hidden cols
// [16w,16w+16) of Hd=256. h in ping-pong LDS; c in registers; f16 weights.
__global__ __launch_bounds__(512, 1)
void enc_layer_kernel(const f16* __restrict__ xin,    // [B,S,512] f16
                      f16* __restrict__ yout,         // [B,S,512] f16
                      const f16* __restrict__ Wih_l,  // layer base [2,1024,512] f16
                      const f16* __restrict__ Whh_l,  // layer base [2,1024,256] f16
                      const float* __restrict__ bih_l,// [2,1024]
                      const float* __restrict__ bhh_l,// [2,1024]
                      f16* __restrict__ hT_out,       // [B,512] (layer slice)
                      float* __restrict__ cT_out) {   // [B,512] (layer slice)
  const int dir  = blockIdx.x;
  const int tid  = threadIdx.x;
  const int wave = tid >> 5, lane = tid & 31;
  const f16* Wih = Wih_l + (long)dir * 1024 * EE;
  const f16* Whh = Whh_l + (long)dir * 1024 * HD;

  __shared__ f16 hbuf[2][BB * HD];  // 2 x 16KB ping-pong hidden state
  for (int i = tid; i < BB * HD; i += 512) hbuf[0][i] = (f16)0.f;
  __syncthreads();

  const int col = (wave << 4) + (lane & 15);
  float creg[2][8];
#pragma unroll
  for (int mt = 0; mt < 2; ++mt)
#pragma unroll
    for (int r = 0; r < 8; ++r) creg[mt][r] = 0.f;

  float bias[4];
#pragma unroll
  for (int g = 0; g < 4; ++g) {
    int n = g * HD + col;
    bias[g] = bih_l[dir * 1024 + n] + bhh_l[dir * 1024 + n];
  }

  int cur = 0;
  for (int step = 0; step < SS; ++step) {
    const int t = dir ? (SS - 1 - step) : step;
    const f16* xt = xin + (long)t * EE;  // row stride = S*E

    v8f acc[2][4];
#pragma unroll
    for (int mt = 0; mt < 2; ++mt)
#pragma unroll
      for (int g = 0; g < 4; ++g) acc[mt][g] = vzero8();

    // gates += x_t @ Wih.T   (K = 512)
    for (int k = 0; k < EE; k += 32) {
      v16h a0 = load_a16(xt, (long)SS * EE, 0, k, lane);
      v16h a1 = load_a16(xt, (long)SS * EE, 16, k, lane);
#pragma unroll
      for (int g = 0; g < 4; ++g) {
        v16h b = load_b_h16(Wih, EE, g * HD + (wave << 4), k, lane);
        acc[0][g] = wmma_f16(a0, b, acc[0][g]);
        acc[1][g] = wmma_f16(a1, b, acc[1][g]);
      }
    }
    // gates += h @ Whh.T     (K = 256, h from LDS)
    for (int k = 0; k < HD; k += 32) {
      v16h a0 = load_a16(&hbuf[cur][0], HD, 0, k, lane);
      v16h a1 = load_a16(&hbuf[cur][0], HD, 16, k, lane);
#pragma unroll
      for (int g = 0; g < 4; ++g) {
        v16h b = load_b_h16(Whh, HD, g * HD + (wave << 4), k, lane);
        acc[0][g] = wmma_f16(a0, b, acc[0][g]);
        acc[1][g] = wmma_f16(a1, b, acc[1][g]);
      }
    }
    // LSTM cell (lane-local: D-tile lane holds col=lane&15, rows r+8*(lane>>4))
#pragma unroll
    for (int mt = 0; mt < 2; ++mt) {
#pragma unroll
      for (int r = 0; r < 8; ++r) {
        float iv = sigmoidf_(acc[mt][0][r] + bias[0]);
        float fv = sigmoidf_(acc[mt][1][r] + bias[1]);
        float gv = tanhf(acc[mt][2][r] + bias[2]);
        float ov = sigmoidf_(acc[mt][3][r] + bias[3]);
        float c  = fv * creg[mt][r] + iv * gv;
        creg[mt][r] = c;
        float h = ov * tanhf(c);
        int m = mt * 16 + ((lane >> 4) << 3) + r;
        hbuf[cur ^ 1][m * HD + col] = (f16)h;
        yout[(long)m * (SS * 512) + (long)t * 512 + dir * HD + col] = (f16)h;
      }
    }
    __syncthreads();
    cur ^= 1;
  }
  // final (hT, cT) -> decoder init slices
#pragma unroll
  for (int mt = 0; mt < 2; ++mt)
#pragma unroll
    for (int r = 0; r < 8; ++r) {
      int m = mt * 16 + ((lane >> 4) << 3) + r;
      cT_out[(long)m * HH + dir * HD + col] = creg[mt][r];
      hT_out[(long)m * HH + dir * HD + col] = hbuf[cur][m * HD + col];
    }
}

// ---- decoder step init: xt = [emb(t=0) | zeros] ----
__global__ void dec_init_kernel(const int* __restrict__ tgt,
                                const float* __restrict__ emb_w,
                                f16* __restrict__ xt) {
  int i = blockIdx.x * 256 + threadIdx.x;  // < B*1024
  int b = i >> 10, c = i & 1023;
  if (c < 512) {
    int tok = tgt[b * TT + 0];
    xt[i] = (f16)emb_w[(long)tok * EE + c];
  } else {
    xt[i] = (f16)0.f;  // h_tilde(0) = 0
  }
}

// ---- one decoder LSTM layer for one timestep ----
// 4 blocks x 256 threads = 32 waves (2/SIMD); wave = hidden slice (16 cols),
// owns all 4 gate tiles x 2 M-tiles. Software-pipelined K loops (double-
// buffered fragments) so next-k loads overlap current-k wmmas.
__global__ __launch_bounds__(256, 1)
void dec_layer_kernel(const f16* __restrict__ xin, long ldx, int Kin,
                      const f16* __restrict__ Wi,   // [2048, Kin] f16
                      const f16* __restrict__ Wh,   // [2048, 512] f16
                      const float* __restrict__ b1, const float* __restrict__ b2,
                      const f16* __restrict__ hOld,   // [B,H] f16 (prev step)
                      f16* __restrict__ hNew,         // [B,H] f16 (this step)
                      float* __restrict__ cL) {       // [B,H] f32
  const int wave = threadIdx.x >> 5, lane = threadIdx.x & 31;
  const int nbase = (blockIdx.x * 8 + wave) << 4;  // hidden col base 0..496

  v8f acc[2][4];
#pragma unroll
  for (int mt = 0; mt < 2; ++mt)
#pragma unroll
    for (int g = 0; g < 4; ++g) acc[mt][g] = vzero8();

  // --- x @ Wi.T (pipelined) ---
  {
    v16h a0 = load_a16(xin, ldx, 0, 0, lane);
    v16h a1 = load_a16(xin, ldx, 16, 0, lane);
    v16h b0 = load_b_h16(Wi, Kin, 0 * HH + nbase, 0, lane);
    v16h b1f = load_b_h16(Wi, Kin, 1 * HH + nbase, 0, lane);
    v16h b2f = load_b_h16(Wi, Kin, 2 * HH + nbase, 0, lane);
    v16h b3f = load_b_h16(Wi, Kin, 3 * HH + nbase, 0, lane);
    for (int k = 0; k < Kin; k += 32) {
      const int kn = (k + 32 < Kin) ? k + 32 : 0;  // clamped prefetch
      v16h na0 = load_a16(xin, ldx, 0, kn, lane);
      v16h na1 = load_a16(xin, ldx, 16, kn, lane);
      v16h nb0 = load_b_h16(Wi, Kin, 0 * HH + nbase, kn, lane);
      v16h nb1 = load_b_h16(Wi, Kin, 1 * HH + nbase, kn, lane);
      v16h nb2 = load_b_h16(Wi, Kin, 2 * HH + nbase, kn, lane);
      v16h nb3 = load_b_h16(Wi, Kin, 3 * HH + nbase, kn, lane);
      acc[0][0] = wmma_f16(a0, b0, acc[0][0]);
      acc[1][0] = wmma_f16(a1, b0, acc[1][0]);
      acc[0][1] = wmma_f16(a0, b1f, acc[0][1]);
      acc[1][1] = wmma_f16(a1, b1f, acc[1][1]);
      acc[0][2] = wmma_f16(a0, b2f, acc[0][2]);
      acc[1][2] = wmma_f16(a1, b2f, acc[1][2]);
      acc[0][3] = wmma_f16(a0, b3f, acc[0][3]);
      acc[1][3] = wmma_f16(a1, b3f, acc[1][3]);
      a0 = na0; a1 = na1; b0 = nb0; b1f = nb1; b2f = nb2; b3f = nb3;
    }
  }
  // --- h_prev @ Wh.T (pipelined, K = 512) ---
  {
    v16h a0 = load_a16(hOld, HH, 0, 0, lane);
    v16h a1 = load_a16(hOld, HH, 16, 0, lane);
    v16h b0 = load_b_h16(Wh, HH, 0 * HH + nbase, 0, lane);
    v16h b1f = load_b_h16(Wh, HH, 1 * HH + nbase, 0, lane);
    v16h b2f = load_b_h16(Wh, HH, 2 * HH + nbase, 0, lane);
    v16h b3f = load_b_h16(Wh, HH, 3 * HH + nbase, 0, lane);
    for (int k = 0; k < HH; k += 32) {
      const int kn = (k + 32 < HH) ? k + 32 : 0;
      v16h na0 = load_a16(hOld, HH, 0, kn, lane);
      v16h na1 = load_a16(hOld, HH, 16, kn, lane);
      v16h nb0 = load_b_h16(Wh, HH, 0 * HH + nbase, kn, lane);
      v16h nb1 = load_b_h16(Wh, HH, 1 * HH + nbase, kn, lane);
      v16h nb2 = load_b_h16(Wh, HH, 2 * HH + nbase, kn, lane);
      v16h nb3 = load_b_h16(Wh, HH, 3 * HH + nbase, kn, lane);
      acc[0][0] = wmma_f16(a0, b0, acc[0][0]);
      acc[1][0] = wmma_f16(a1, b0, acc[1][0]);
      acc[0][1] = wmma_f16(a0, b1f, acc[0][1]);
      acc[1][1] = wmma_f16(a1, b1f, acc[1][1]);
      acc[0][2] = wmma_f16(a0, b2f, acc[0][2]);
      acc[1][2] = wmma_f16(a1, b2f, acc[1][2]);
      acc[0][3] = wmma_f16(a0, b3f, acc[0][3]);
      acc[1][3] = wmma_f16(a1, b3f, acc[1][3]);
      a0 = na0; a1 = na1; b0 = nb0; b1f = nb1; b2f = nb2; b3f = nb3;
    }
  }

  const int coln = nbase + (lane & 15);
  float bsum[4];
#pragma unroll
  for (int g = 0; g < 4; ++g) bsum[g] = b1[g * HH + coln] + b2[g * HH + coln];
#pragma unroll
  for (int mt = 0; mt < 2; ++mt)
#pragma unroll
    for (int r = 0; r < 8; ++r) {
      int m = mt * 16 + ((lane >> 4) << 3) + r;
      long idx = (long)m * HH + coln;
      float iv = sigmoidf_(acc[mt][0][r] + bsum[0]);
      float fv = sigmoidf_(acc[mt][1][r] + bsum[1]);
      float gv = tanhf(acc[mt][2][r] + bsum[2]);
      float ov = sigmoidf_(acc[mt][3][r] + bsum[3]);
      float c  = fv * cL[idx] + iv * gv;
      cL[idx] = c;
      hNew[idx] = (f16)(ov * tanhf(c));
    }
}

// ---- q = dec_out @ attn_W.T (f32 out); pipelined ----
__global__ __launch_bounds__(256, 1)
void dec_q_kernel(const f16* __restrict__ dec_out, const f16* __restrict__ attn_W16,
                  float* __restrict__ qbuf) {
  const int wave = threadIdx.x >> 5, lane = threadIdx.x & 31;
  const int nbase = (blockIdx.x * 8 + wave) << 4;
  v8f q0 = vzero8(), q1 = vzero8();
  v16h a0 = load_a16(dec_out, HH, 0, 0, lane);
  v16h a1 = load_a16(dec_out, HH, 16, 0, lane);
  v16h b  = load_b_h16(attn_W16, HH, nbase, 0, lane);
  for (int k = 0; k < HH; k += 32) {
    const int kn = (k + 32 < HH) ? k + 32 : 0;
    v16h na0 = load_a16(dec_out, HH, 0, kn, lane);
    v16h na1 = load_a16(dec_out, HH, 16, kn, lane);
    v16h nb  = load_b_h16(attn_W16, HH, nbase, kn, lane);
    q0 = wmma_f16(a0, b, q0);
    q1 = wmma_f16(a1, b, q1);
    a0 = na0; a1 = na1; b = nb;
  }
  const int coln = nbase + (lane & 15);
#pragma unroll
  for (int r = 0; r < 8; ++r) {
    int hi8 = ((lane >> 4) << 3) + r;
    qbuf[(long)hi8 * HH + coln]        = q0[r];
    qbuf[(long)(16 + hi8) * HH + coln] = q1[r];
  }
}

// ---- attention: one block per batch row; scores + softmax + ctx in LDS ----
__global__ __launch_bounds__(256, 1)
void dec_attn_kernel(const f16* __restrict__ h_src, const float* __restrict__ qbuf,
                     const f16* __restrict__ dec_out, f16* __restrict__ cc) {
  __shared__ float ssc[SS];
  __shared__ float red[256];
  const int b = blockIdx.x, tid = threadIdx.x;

  // score for s = tid
  const f16* hs = h_src + ((long)b * SS + tid) * HH;
  const float* q = qbuf + (long)b * HH;
  float a = 0.f;
  for (int h = 0; h < HH; ++h) a += (float)hs[h] * q[h];

  red[tid] = a; __syncthreads();
  for (int s = 128; s; s >>= 1) {
    if (tid < s) red[tid] = fmaxf(red[tid], red[tid + s]);
    __syncthreads();
  }
  float mx = red[0]; __syncthreads();
  float e = __expf(a - mx);
  ssc[tid] = e;
  red[tid] = e; __syncthreads();
  for (int s = 128; s; s >>= 1) {
    if (tid < s) red[tid] += red[tid + s];
    __syncthreads();
  }
  const float inv = 1.f / red[0];

  // ctx[b,h] and cc = [dec_out | ctx]
  for (int h = tid; h < HH; h += 256) {
    const f16* hp = h_src + (long)b * SS * HH + h;
    float av = 0.f;
    for (int s = 0; s < SS; ++s) av += ssc[s] * (float)hp[(long)s * HH];
    cc[b * 1024 + 512 + h] = (f16)(av * inv);
    cc[b * 1024 + h] = dec_out[(long)b * HH + h];
  }
}

// ---- h_tilde = tanh(cc @ concat_W.T + b); also prefetch next-step embedding ----
__global__ __launch_bounds__(256, 1)
void dec_htilde_kernel(const f16* __restrict__ cc, const f16* __restrict__ cat16,
                       const float* __restrict__ catb, f16* __restrict__ xt,
                       f16* __restrict__ h_last, int t,
                       const int* __restrict__ tgt, const float* __restrict__ emb_w) {
  const int wave = threadIdx.x >> 5, lane = threadIdx.x & 31;
  const int nbase = (blockIdx.x * 8 + wave) << 4;
  v8f d0 = vzero8(), d1 = vzero8();
  v16h a0 = load_a16(cc, 1024, 0, 0, lane);
  v16h a1 = load_a16(cc, 1024, 16, 0, lane);
  v16h b  = load_b_h16(cat16, 1024, nbase, 0, lane);
  for (int k = 0; k < 1024; k += 32) {
    const int kn = (k + 32 < 1024) ? k + 32 : 0;
    v16h na0 = load_a16(cc, 1024, 0, kn, lane);
    v16h na1 = load_a16(cc, 1024, 16, kn, lane);
    v16h nb  = load_b_h16(cat16, 1024, nbase, kn, lane);
    d0 = wmma_f16(a0, b, d0);
    d1 = wmma_f16(a1, b, d1);
    a0 = na0; a1 = na1; b = nb;
  }
  const int coln = nbase + (lane & 15);
  const float bb = catb[coln];
#pragma unroll
  for (int r = 0; r < 8; ++r) {
    int hi8 = ((lane >> 4) << 3) + r;
    f16 v0 = (f16)tanhf(d0[r] + bb);
    f16 v1 = (f16)tanhf(d1[r] + bb);
    xt[hi8 * 1024 + 512 + coln]        = v0;
    xt[(16 + hi8) * 1024 + 512 + coln] = v1;
    if (t == TT - 1) {
      h_last[(long)hi8 * HH + coln]        = v0;
      h_last[(long)(16 + hi8) * HH + coln] = v1;
    }
  }
  // gather embedding for step t+1 into xt[:, :512]
  if (t + 1 < TT) {
    int gtid = blockIdx.x * 256 + threadIdx.x;  // 1024 threads total
    for (int i = gtid; i < BB * EE; i += 1024) {
      int b2 = i >> 9, e = i & 511;
      int tok = tgt[b2 * TT + t + 1];
      xt[b2 * 1024 + e] = (f16)emb_w[(long)tok * EE + e];
    }
  }
}

// ---- generator: logits[32,32000] = h_last @ gen_W.T + b (wide WMMA GEMM) ----
__global__ __launch_bounds__(256)
void gen_kernel(const f16* __restrict__ h_last, const float* __restrict__ gen_W,
                const float* __restrict__ gen_b, float* __restrict__ logits) {
  const int wave = threadIdx.x >> 5, lane = threadIdx.x & 31;
  const int n0 = (blockIdx.x * 8 + wave) << 4;  // 2000 n-tiles
  v8f d0 = vzero8(), d1 = vzero8();
  for (int k = 0; k < HH; k += 32) {
    v16h a0 = load_a16(h_last, HH, 0, k, lane);
    v16h a1 = load_a16(h_last, HH, 16, k, lane);
    v16h b  = load_b_w32(gen_W, HH, n0, k, lane);
    d0 = wmma_f16(a0, b, d0);
    d1 = wmma_f16(a1, b, d1);
  }
  const int col = n0 + (lane & 15);
  const float bb = gen_b[col];
#pragma unroll
  for (int r = 0; r < 8; ++r) {
    int hi8 = ((lane >> 4) << 3) + r;
    logits[(long)hi8 * VV + col]        = d0[r] + bb;
    logits[(long)(16 + hi8) * VV + col] = d1[r] + bb;
  }
}

// ---- softmax over V=32000 per row; writes final y_hat ----
__global__ __launch_bounds__(256)
void softmax_v_kernel(const float* __restrict__ logits, float* __restrict__ out) {
  __shared__ float red[256];
  const int b = blockIdx.x, tid = threadIdx.x;
  const float* row = logits + (long)b * VV;
  float mx = -1e30f;
  for (int v = tid; v < VV; v += 256) mx = fmaxf(mx, row[v]);
  red[tid] = mx; __syncthreads();
  for (int s = 128; s; s >>= 1) { if (tid < s) red[tid] = fmaxf(red[tid], red[tid + s]); __syncthreads(); }
  mx = red[0]; __syncthreads();
  float sum = 0.f;
  for (int v = tid; v < VV; v += 256) sum += __expf(row[v] - mx);
  red[tid] = sum; __syncthreads();
  for (int s = 128; s; s >>= 1) { if (tid < s) red[tid] += red[tid + s]; __syncthreads(); }
  const float inv = 1.f / red[0];
  for (int v = tid; v < VV; v += 256) out[(long)b * VV + v] = __expf(row[v] - mx) * inv;
}

extern "C" void kernel_launch(void* const* d_in, const int* in_sizes, int n_in,
                              void* d_out, int out_size, void* d_ws, size_t ws_size,
                              hipStream_t stream) {
  (void)in_sizes; (void)n_in; (void)out_size; (void)ws_size;
  const int*   src   = (const int*)d_in[0];
  const int*   tgt   = (const int*)d_in[1];
  const float* emb_s = (const float*)d_in[2];
  const float* emb_d = (const float*)d_in[3];
  const float* eWih  = (const float*)d_in[4];
  const float* eWhh  = (const float*)d_in[5];
  const float* ebih  = (const float*)d_in[6];
  const float* ebhh  = (const float*)d_in[7];
  const float* dWih0 = (const float*)d_in[8];
  const float* dWihR = (const float*)d_in[9];
  const float* dWhh  = (const float*)d_in[10];
  const float* dbih  = (const float*)d_in[11];
  const float* dbhh  = (const float*)d_in[12];
  const float* attnW = (const float*)d_in[13];
  const float* catW  = (const float*)d_in[14];
  const float* catb  = (const float*)d_in[15];
  const float* genW  = (const float*)d_in[16];
  const float* genb  = (const float*)d_in[17];
  float* out = (float*)d_out;

  char* ws = (char*)d_ws;                        // workspace layout (~52 MB)
  f16*   xp   = (f16*)(ws + 0);                  // ping [B,S,512] f16    8.0 MB
  f16*   xq   = (f16*)(ws + 8388608);            // pong                  8.0 MB
  f16*   hA   = (f16*)(ws + 16777216);           // [4,B,H] f16    128 KB
  f16*   hB   = (f16*)(ws + 16908288);           // [4,B,H] f16    128 KB
  float* cSt  = (float*)(ws + 17039360);         // [4,B,H] f32    256 KB
  f16*   xtb  = (f16*)(ws + 17301504);           // [B,1024] f16    64 KB
  float* qb   = (float*)(ws + 17367040);         // [B,H] f32       64 KB
  f16*   ccb  = (f16*)(ws + 17432576);           // [B,1024] f16    64 KB
  f16*   hl   = (f16*)(ws + 17498112);           // [B,H] f16       32 KB
  float* lg   = (float*)(ws + 17530880);         // [B,V] f32      3.9 MB
  f16* We16ih = (f16*)(ws + 21626880);           // [4,2,1024,512]  8.4 MB
  f16* We16hh = (f16*)(ws + 30015488);           // [4,2,1024,256]  4.2 MB
  f16* Wd16i0 = (f16*)(ws + 34209792);           // [2048,1024]     4.2 MB
  f16* Wd16iR = (f16*)(ws + 38404096);           // [3,2048,512]    6.3 MB
  f16* Wd16hh = (f16*)(ws + 44695552);           // [4,2048,512]    8.4 MB
  f16* attn16 = (f16*)(ws + 53084160);           // [512,512]       0.5 MB
  f16* cat16  = (f16*)(ws + 53608448);           // [512,1024]      1.0 MB

  // ---- pre-convert recurrent weights to f16 (halves L2 stream in the scans) ----
  cvt_f32_f16_kernel<<<2048, 256, 0, stream>>>(eWih,  We16ih, 4 * 2 * 1024 * EE);
  cvt_f32_f16_kernel<<<2048, 256, 0, stream>>>(eWhh,  We16hh, 4 * 2 * 1024 * HD);
  cvt_f32_f16_kernel<<<2048, 256, 0, stream>>>(dWih0, Wd16i0, 2048 * 1024);
  cvt_f32_f16_kernel<<<2048, 256, 0, stream>>>(dWihR, Wd16iR, 3 * 2048 * HH);
  cvt_f32_f16_kernel<<<2048, 256, 0, stream>>>(dWhh,  Wd16hh, 4 * 2048 * HH);
  cvt_f32_f16_kernel<<<512, 256, 0, stream>>>(attnW, attn16, HH * HH);
  cvt_f32_f16_kernel<<<512, 256, 0, stream>>>(catW,  cat16,  HH * 1024);

  // ---- encoder ----
  const int nemb = BB * SS * EE;
  embed_src_kernel<<<(nemb + 255) / 256, 256, 0, stream>>>(src, emb_s, xp, nemb);
  for (int l = 0; l < 4; ++l) {
    const f16* inb = (l & 1) ? xq : xp;
    f16* outb      = (l & 1) ? xp : xq;
    enc_layer_kernel<<<2, 512, 0, stream>>>(
        inb, outb,
        We16ih + (long)l * 2 * 1024 * EE, We16hh + (long)l * 2 * 1024 * HD,
        ebih + (long)l * 2 * 1024, ebhh + (long)l * 2 * 1024,
        hA + (long)l * BB * HH, cSt + (long)l * BB * HH);
  }
  // after l=3: in=xq, out=xp -> h_src = xp

  // ---- decoder: per-timestep launches, h-state ping-pong by pointer swap ----
  dec_init_kernel<<<(BB * 1024) / 256, 256, 0, stream>>>(tgt, emb_d, xtb);
  f16* hPrev = hA;
  f16* hNew  = hB;
  for (int t = 0; t < TT; ++t) {
    for (int l = 0; l < 4; ++l) {
      const f16* xin = (l == 0) ? xtb : (hNew + (long)(l - 1) * BB * HH);
      const long ldx = (l == 0) ? 1024 : HH;
      const int  Kin = (l == 0) ? 1024 : HH;
      const f16* Wi  = (l == 0) ? Wd16i0 : (Wd16iR + (long)(l - 1) * 2048 * HH);
      dec_layer_kernel<<<4, 256, 0, stream>>>(
          xin, ldx, Kin, Wi, Wd16hh + (long)l * 2048 * HH,
          dbih + l * 2048, dbhh + l * 2048,
          hPrev + (long)l * BB * HH, hNew + (long)l * BB * HH,
          cSt + (long)l * BB * HH);
    }
    const f16* dec_out = hNew + (long)3 * BB * HH;
    dec_q_kernel<<<4, 256, 0, stream>>>(dec_out, attn16, qb);
    dec_attn_kernel<<<32, 256, 0, stream>>>(xp, qb, dec_out, ccb);
    dec_htilde_kernel<<<4, 256, 0, stream>>>(ccb, cat16, catb, xtb, hl, t, tgt,
                                             emb_d);
    f16* tmp = hPrev; hPrev = hNew; hNew = tmp;
  }

  // ---- generator + vocab softmax ----
  gen_kernel<<<250, 256, 0, stream>>>(hl, genW, genb, lg);
  softmax_v_kernel<<<32, 256, 0, stream>>>(lg, out);
}
